// DeformableDynamicGather2D_27736898797670
// MI455X (gfx1250) — matrix operand order repaired and checked by reading, compiled, not run on hardware
//
#include <hip/hip_runtime.h>
#include <hip/hip_bf16.h>
#include <math.h>

typedef __attribute__((ext_vector_type(16))) _Float16 v16h;
typedef __attribute__((ext_vector_type(8)))  float    v8f;

// Problem constants (fixed by the reference)
#define BB     4
#define CCH    64
#define HH     256
#define WW     256
#define NP     16384
#define HW     (HH*WW)        // 65536
#define HID    64
#define K1PAD  96             // 68 -> 96 (3 x K=32)
#define OPAD   48             // 33 -> 48 (3 x N=16)
#define KDEF   9

__device__ __forceinline__ float softplus_f(float x) {
    return x > 20.f ? x : log1pf(expf(x));
}
__device__ __forceinline__ float leaky_f(float x) {
    return x >= 0.f ? x : 0.2f * x;
}

// Bilinear setup: grid coords in [-1,1] -> corner indices + weights
__device__ __forceinline__ void bilin_setup(float gx, float gy,
                                            int& i00, int& i01, int& i10, int& i11,
                                            float& w00, float& w01, float& w10, float& w11) {
    float ix = fminf(fmaxf((gx + 1.f) * 0.5f * (float)(WW - 1), 0.f), (float)(WW - 1));
    float iy = fminf(fmaxf((gy + 1.f) * 0.5f * (float)(HH - 1), 0.f), (float)(HH - 1));
    float fx0 = floorf(ix), fy0 = floorf(iy);
    float wx = ix - fx0, wy = iy - fy0;
    int x0 = (int)fx0, y0 = (int)fy0;
    int x1 = min(x0 + 1, WW - 1), y1 = min(y0 + 1, HH - 1);
    i00 = y0 * WW + x0; i01 = y0 * WW + x1;
    i10 = y1 * WW + x0; i11 = y1 * WW + x1;
    w00 = (1.f - wx) * (1.f - wy); w01 = wx * (1.f - wy);
    w10 = (1.f - wx) * wy;         w11 = wx * wy;
}

// ---------------------------------------------------------------------------
// Fragment layouts (wave32 WMMA, f16 16x16x32):
//   A-frag: lane l, element e holds A[row = l&15][K = kb + (l>>4)*8 + (e&7) + ((e&8)<<1)]
//   B-frag: lane l, element e holds B[K = kb + (l>>4)*8 + (e&7) + ((e&8)<<1)][N = t*16 + (l&15)]
//   C/D:    lane l, vgpr r  holds D[M = r + (l>>4)*8][N = t*16 + (l&15)]
// All LDS operands are stored PRE-SWIZZLED so each lane's 16 halves are 32
// contiguous bytes -> v16h loads lower to 2x ds_load_b128.
// Inverse map (given k-remainder kr = K % 32):
//   holder half  hh = (kr>>3)&1
//   element      e  = (kr&7) | ((kr>>4)<<3)
// ---------------------------------------------------------------------------

__global__ __launch_bounds__(128)
void deform_gather_fused(const float* __restrict__ feat,
                         const float* __restrict__ coords,
                         const float* __restrict__ cell,
                         const float* __restrict__ W1,
                         const float* __restrict__ b1,
                         const float* __restrict__ Wr,
                         const float* __restrict__ br,
                         const float* __restrict__ W2,
                         const float* __restrict__ b2,
                         float* __restrict__ outp) {
    // ---- LDS (everything in fragment-swizzled order) ----
    __shared__ alignas(32) _Float16 w1swz[3 * 4 * 32 * 16];   // [ks][t][lane][e]
    __shared__ alignas(32) _Float16 wrswz[2 * 4 * 32 * 16];
    __shared__ alignas(32) _Float16 w2swz[2 * 3 * 32 * 16];
    __shared__ alignas(32) _Float16 aswz[4][3 * 32 * 16];     // per-wave A tile
    __shared__ alignas(32) _Float16 hswz[4][2 * 32 * 16];     // per-wave hidden
    __shared__ float b1s[HID], brs[HID], b2s[OPAD];
    __shared__ float os[4][16][OPAD];                          // per-wave MLP out

    const int tid  = threadIdx.x;
    const int wave = tid >> 5;
    const int lane = tid & 31;
    const int p    = lane & 15;   // point row within wave tile (M)
    const int hsel = lane >> 4;   // half-wave selector

    // ---- Phase 0: stage weights into LDS, pre-swizzled to B-frag order ----
    for (int i = tid; i < 3 * 4 * 32 * 16; i += 128) {
        int e = i & 15, ln = (i >> 4) & 31, t = (i >> 9) & 3, ks = i >> 11;
        int k = ks * 32 + (ln >> 4) * 8 + (e & 7) + ((e >> 3) << 4);
        int n = t * 16 + (ln & 15);
        w1swz[i] = (_Float16)((k < CCH + 4) ? W1[k * HID + n] : 0.f);
    }
    for (int i = tid; i < 2 * 4 * 32 * 16; i += 128) {
        int e = i & 15, ln = (i >> 4) & 31, t = (i >> 9) & 3, ks = i >> 11;
        int k = ks * 32 + (ln >> 4) * 8 + (e & 7) + ((e >> 3) << 4);
        int n = t * 16 + (ln & 15);
        wrswz[i] = (_Float16)Wr[k * HID + n];
    }
    for (int i = tid; i < 2 * 3 * 32 * 16; i += 128) {
        int e = i & 15, ln = (i >> 4) & 31;
        int rest = i >> 9;                       // 0..5
        int t = rest % 3, ks = rest / 3;
        int k = ks * 32 + (ln >> 4) * 8 + (e & 7) + ((e >> 3) << 4);
        int n = t * 16 + (ln & 15);
        w2swz[i] = (_Float16)((n < 33) ? W2[k * 33 + n] : 0.f);
    }
    if (tid < HID) { b1s[tid] = b1[tid]; brs[tid] = br[tid]; }
    if (tid < OPAD) b2s[tid] = (tid < 33) ? b2[tid] : 0.f;

    // ---- point identity ----
    const int g = blockIdx.x * 64 + wave * 16 + p;   // global point, exact cover
    const int b = g >> 14;                            // / 16384
    const float cx  = coords[2 * g + 0];
    const float cy  = coords[2 * g + 1];
    const float clx = cell[2 * g + 0];
    const float cly = cell[2 * g + 1];
    const float* featb = feat + (size_t)b * CCH * HW;

    __syncthreads();

    // ---- Phase 1: anchor bilinear gather -> A tile (fragment order) ----
    {
        int i00, i01, i10, i11; float w00, w01, w10, w11;
        bilin_setup(cx, cy, i00, i01, i10, i11, w00, w01, w10, w11);

        // Prefetch the deform-phase cache lines for this lane's channel half
        // while the MLP runs (emits global_prefetch_b8).
        const float* fpre = featb + (size_t)(hsel * 32) * HW;
        #pragma unroll
        for (int c = 0; c < 32; c += 4)
            __builtin_prefetch(fpre + (size_t)c * HW + i00, 0, 0);

        _Float16* aw = aswz[wave];
        for (int c = hsel; c < K1PAD; c += 2) {
            float v;
            if (c < CCH) {
                const float* fc = featb + (size_t)c * HW;
                v = fc[i00] * w00 + fc[i01] * w01 + fc[i10] * w10 + fc[i11] * w11;
            } else if (c == 64) v = cx;
            else if (c == 65)   v = cy;
            else if (c == 66)   v = clx;
            else if (c == 67)   v = cly;
            else                v = 0.f;
            int ks = c >> 5, kr = c & 31;
            int hh = (kr >> 3) & 1;
            int e  = (kr & 7) | ((kr >> 4) << 3);
            int ln = p | (hh << 4);
            aw[(ks * 32 + ln) * 16 + e] = (_Float16)v;
        }
    }
    __syncthreads();

    v8f acc[4];
    #pragma unroll
    for (int t = 0; t < 4; t++) acc[t] = (v8f){};

    // ---- Phase 2: GEMM1  h1 = leaky(rin @ W1 + b1), K=96 ----
    #pragma unroll
    for (int ks = 0; ks < 3; ks++) {
        v16h af = *(const v16h*)(aswz[wave] + (ks * 32 + lane) * 16);
        #pragma unroll
        for (int t = 0; t < 4; t++) {
            v16h bf = *(const v16h*)(w1swz + ((ks * 4 + t) * 32 + lane) * 16);
            acc[t] = __builtin_amdgcn_wmma_f32_16x16x32_f16(
                false, af, false, bf, (short)0, acc[t], false, false);
        }
    }
    {
        _Float16* hw = hswz[wave];
        #pragma unroll
        for (int t = 0; t < 4; t++) {
            float bb = b1s[t * 16 + p];
            int e      = (p & 7) | ((t & 1) << 3);
            int lnbase = (t >> 1) * 32 + ((p >> 3) << 4) + hsel * 8;
            #pragma unroll
            for (int r = 0; r < 8; r++) {
                float v = leaky_f(acc[t][r] + bb);
                acc[t][r] = v;                         // keep h1 (f32) for residual
                hw[(lnbase + r) * 16 + e] = (_Float16)v;
            }
        }
    }
    __syncthreads();

    // ---- Phase 3: GEMM2  h = leaky(h1 + h1 @ Wr + br), K=64 ----
    // residual add is free: accumulator already holds h1
    #pragma unroll
    for (int ks = 0; ks < 2; ks++) {
        v16h af = *(const v16h*)(hswz[wave] + (ks * 32 + lane) * 16);
        #pragma unroll
        for (int t = 0; t < 4; t++) {
            v16h bf = *(const v16h*)(wrswz + ((ks * 4 + t) * 32 + lane) * 16);
            acc[t] = __builtin_amdgcn_wmma_f32_16x16x32_f16(
                false, af, false, bf, (short)0, acc[t], false, false);
        }
    }
    __syncthreads();   // all reads of hswz done before overwrite
    {
        _Float16* hw = hswz[wave];
        #pragma unroll
        for (int t = 0; t < 4; t++) {
            float bb = brs[t * 16 + p];
            int e      = (p & 7) | ((t & 1) << 3);
            int lnbase = (t >> 1) * 32 + ((p >> 3) << 4) + hsel * 8;
            #pragma unroll
            for (int r = 0; r < 8; r++) {
                float v = leaky_f(acc[t][r] + bb);
                hw[(lnbase + r) * 16 + e] = (_Float16)v;
            }
        }
    }
    __syncthreads();

    // ---- Phase 4: GEMM3  out = h @ W2 + b2, K=64, N padded to 48 ----
    v8f oc[3];
    #pragma unroll
    for (int t = 0; t < 3; t++) oc[t] = (v8f){};
    #pragma unroll
    for (int ks = 0; ks < 2; ks++) {
        v16h af = *(const v16h*)(hswz[wave] + (ks * 32 + lane) * 16);
        #pragma unroll
        for (int t = 0; t < 3; t++) {
            v16h bf = *(const v16h*)(w2swz + ((ks * 3 + t) * 32 + lane) * 16);
            oc[t] = __builtin_amdgcn_wmma_f32_16x16x32_f16(
                false, af, false, bf, (short)0, oc[t], false, false);
        }
    }
    #pragma unroll
    for (int t = 0; t < 3; t++) {
        float bb = b2s[t * 16 + p];
        #pragma unroll
        for (int r = 0; r < 8; r++)
            os[wave][r + hsel * 8][t * 16 + p] = oc[t][r] + bb;
    }
    __syncthreads();

    // ---- Phase 5: deformable gather + weighted reduce ----
    const float* orow = os[wave][p];
    float r_act = fminf(fmaxf(softplus_f(orow[0]) + 0.1f, 0.1f), 4.0f);
    float sigma = fminf(fmaxf(softplus_f(orow[1]) + 0.5f, 0.5f), 6.0f);
    float sig_eff = sigma * 2.0f;                 // TAU2D
    float denom   = sig_eff * sig_eff + 1e-8f;
    const float scx = 2.f / (float)(WW - 1);
    const float scy = 2.f / (float)(HH - 1);

    float accum[32];
    #pragma unroll
    for (int c = 0; c < 32; c++) accum[c] = 0.f;
    float wsum = 0.f;

    const float* fbase = featb + (size_t)(hsel * 32) * HW;

    for (int k = 0; k < KDEF; k++) {
        float bx = (float)(k % 3 - 1);
        float by = (float)(k / 3 - 1);
        float rx = tanhf(orow[2 + 2 * k]) * 0.5f;
        float ry = tanhf(orow[3 + 2 * k]) * 0.5f;
        float ox = r_act * bx + rx;
        float oy = r_act * by + ry;
        float d2 = ox * ox + oy * oy;
        float wgeo = expf(-0.5f * d2 / denom);
        float gate = 1.f / (1.f + expf(-orow[20 + k]));
        float wk = wgeo * gate;
        wsum += wk;

        float gx = cx + ox * scx;
        float gy = cy + oy * scy;
        int i00, i01, i10, i11; float w00, w01, w10, w11;
        bilin_setup(gx, gy, i00, i01, i10, i11, w00, w01, w10, w11);
        float c00 = wk * w00, c01 = wk * w01, c10 = wk * w10, c11 = wk * w11;
        #pragma unroll
        for (int c = 0; c < 32; c++) {
            const float* fc = fbase + (size_t)c * HW;
            accum[c] += fc[i00] * c00 + fc[i01] * c01 + fc[i10] * c10 + fc[i11] * c11;
        }
    }

    float inv = 1.f / (wsum + 1e-8f);
    float* op = outp + (size_t)g * CCH + hsel * 32;
    #pragma unroll
    for (int c = 0; c < 32; c++) op[c] = accum[c] * inv;
}

extern "C" void kernel_launch(void* const* d_in, const int* in_sizes, int n_in,
                              void* d_out, int out_size, void* d_ws, size_t ws_size,
                              hipStream_t stream) {
    const float* feat   = (const float*)d_in[0];
    const float* coords = (const float*)d_in[1];
    const float* cell   = (const float*)d_in[2];
    const float* W1     = (const float*)d_in[3];
    const float* b1     = (const float*)d_in[4];
    const float* Wr     = (const float*)d_in[5];
    const float* br     = (const float*)d_in[6];
    const float* W2     = (const float*)d_in[7];
    const float* b2     = (const float*)d_in[8];
    float* outp         = (float*)d_out;

    // 65536 points total, 64 points per block (4 waves x 16)
    dim3 grid((BB * NP) / 64);
    dim3 block(128);
    deform_gather_fused<<<grid, block, 0, stream>>>(feat, coords, cell,
                                                    W1, b1, Wr, br, W2, b2, outp);
}
